// SimpleMoELayer_9689446219889
// MI455X (gfx1250) — matrix operand order repaired and checked by reading, not compiled, not run on hardware
//
#include <hip/hip_runtime.h>
#include <cstdint>
#include <cstddef>

#define HIDDEN 2048
#define INTER  5632
#define NEXP   8
#define NTOK   4096

#define BM 128          // block tile M (token slots)
#define BN 64           // block tile N (output features)
#define BK 32           // K step per WMMA
#define LDA 40          // padded LDS row stride (halves): 80B -> bank-conflict-free b128

typedef __attribute__((ext_vector_type(16))) _Float16 v16h;
typedef __attribute__((ext_vector_type(8)))  float    v8f;

union Frag16 { v16h h; uint4 q[2]; };
union Half4  { _Float16 h[4]; uint2 u; };

// ---------------------------------------------------------------------------
// CDNA5 helpers: async global->LDS DMA (ASYNCcnt) with safe fallbacks
// ---------------------------------------------------------------------------
#if defined(__has_builtin)
#if __has_builtin(__builtin_amdgcn_global_load_async_to_lds_b128)
#define HAVE_ASYNC_LDS 1
#endif
#endif

// builtin signature is (v4i AS1*, v4i AS3*, imm offset, imm cpol)
typedef int v4i __attribute__((ext_vector_type(4)));
typedef v4i __attribute__((address_space(1))) v4i_gl;   // global (AS1) v4i
typedef v4i __attribute__((address_space(3))) v4i_ld;   // LDS (AS3) v4i

__device__ __forceinline__ void async_copy_b128(const void* g, void* l) {
#ifdef HAVE_ASYNC_LDS
  __builtin_amdgcn_global_load_async_to_lds_b128((v4i_gl*)g, (v4i_ld*)l, 0, 0);
#else
  *(uint4*)l = *(const uint4*)g;
#endif
}

__device__ __forceinline__ void wait_async0() {
#if defined(__has_builtin) && __has_builtin(__builtin_amdgcn_s_wait_asynccnt)
  __builtin_amdgcn_s_wait_asynccnt(0);
#else
  asm volatile("s_wait_asynccnt 0x0" ::: "memory");
#endif
}

__device__ __forceinline__ float fast_rcp(float d) {
#if defined(__has_builtin) && __has_builtin(__builtin_amdgcn_rcpf)
  return __builtin_amdgcn_rcpf(d);   // v_rcp_f32, avoids IEEE div expansion
#else
  return 1.0f / d;
#endif
}

// ---------------------------------------------------------------------------
// 0) zero output + per-expert counters (d_out / d_ws are poisoned, not zeroed)
// ---------------------------------------------------------------------------
__global__ void moe_zero_kernel(float* __restrict__ out, int* __restrict__ counts) {
  size_t gid = (size_t)blockIdx.x * blockDim.x + threadIdx.x;
  if (gid < NEXP) counts[gid] = 0;
  const size_t n = (size_t)NTOK * HIDDEN;
  const size_t stride = (size_t)gridDim.x * blockDim.x;
  for (size_t i = gid; i < n; i += stride) out[i] = 0.0f;
}

// ---------------------------------------------------------------------------
// 0b) one-shot f32 -> f16 cast of activations; enables async DMA of A tiles
// ---------------------------------------------------------------------------
__global__ void moe_xcast_kernel(const float* __restrict__ x, _Float16* __restrict__ xh) {
  const size_t n4 = (size_t)NTOK * HIDDEN / 4;
  const size_t stride = (size_t)gridDim.x * blockDim.x;
  for (size_t i = (size_t)blockIdx.x * blockDim.x + threadIdx.x; i < n4; i += stride) {
    const float4 f = ((const float4*)x)[i];
    Half4 hv;
    hv.h[0] = (_Float16)f.x; hv.h[1] = (_Float16)f.y;
    hv.h[2] = (_Float16)f.z; hv.h[3] = (_Float16)f.w;
    *(uint2*)&xh[4 * i] = hv.u;
  }
}

// ---------------------------------------------------------------------------
// 1) router: one wave32 per token. 8 expert logits, softmax top-2, renorm,
//    atomic-append (token, coef) into per-expert lists.
// ---------------------------------------------------------------------------
__global__ __launch_bounds__(256)
void moe_router_kernel(const float* __restrict__ x,
                       const float* __restrict__ gate_w,
                       int* __restrict__ counts,
                       int* __restrict__ tok_list,
                       float* __restrict__ coefs) {
  const int wave = threadIdx.x >> 5;
  const int lane = threadIdx.x & 31;
  const int t = blockIdx.x * 8 + wave;
  if (t >= NTOK) return;

  float acc[NEXP];
#pragma unroll
  for (int e = 0; e < NEXP; ++e) acc[e] = 0.0f;

  const float* xr = x + (size_t)t * HIDDEN;
  for (int h = lane; h < HIDDEN; h += 32) {
    const float xv = xr[h];
#pragma unroll
    for (int e = 0; e < NEXP; ++e) acc[e] += xv * gate_w[e * HIDDEN + h];
  }
#pragma unroll
  for (int off = 16; off > 0; off >>= 1) {
#pragma unroll
    for (int e = 0; e < NEXP; ++e) acc[e] += __shfl_xor(acc[e], off, 32);
  }

  if (lane == 0) {
    float m = acc[0];
#pragma unroll
    for (int e = 1; e < NEXP; ++e) m = fmaxf(m, acc[e]);
    float p[NEXP];
#pragma unroll
    for (int e = 0; e < NEXP; ++e) p[e] = __expf(acc[e] - m);
    int i0 = 0;
#pragma unroll
    for (int e = 1; e < NEXP; ++e) if (p[e] > p[i0]) i0 = e;
    int i1 = (i0 == 0) ? 1 : 0;
#pragma unroll
    for (int e = 0; e < NEXP; ++e) if (e != i0 && p[e] > p[i1]) i1 = e;
    const float w0 = p[i0] / (p[i0] + p[i1]);   // softmax denom cancels
    const float w1 = 1.0f - w0;
    int s0 = atomicAdd(&counts[i0], 1);
    tok_list[i0 * NTOK + s0] = t;
    coefs[i0 * NTOK + s0] = w0;
    int s1 = atomicAdd(&counts[i1], 1);
    tok_list[i1 * NTOK + s1] = t;
    coefs[i1 * NTOK + s1] = w1;
  }
}

// ---------------------------------------------------------------------------
// 2) per-expert gate+up GEMM fused with SiLU:  h = silu(Xe*Gp^T) * (Xe*Up^T)
//    A: token-gathered f16 rows via async global->LDS DMA, double buffered.
//    B: f32 weights -> f16 through VGPRs, register-prefetched, double buffered.
//    Block tile 128x64, 8 waves each owning 32x32 of BOTH accumulators.
// ---------------------------------------------------------------------------
__global__ __launch_bounds__(256)
void moe_ffn_kernel(const _Float16* __restrict__ xh,
                    const float* __restrict__ gp_w,
                    const float* __restrict__ up_w,
                    const int* __restrict__ counts,
                    const int* __restrict__ tok,     // already offset by expert
                    _Float16* __restrict__ hbuf,
                    int e) {
  const int count = counts[e];
  const int mBase = blockIdx.y * BM;
  if (mBase >= count) return;
  const int nBase = blockIdx.x * BN;

  const int tid  = threadIdx.x;
  const int lane = tid & 31;
  const int w    = tid >> 5;
  const int wm   = w & 3;   // 4 waves along M (4*32 = 128)
  const int wn   = w >> 2;  // 2 waves along N (2*32 = 64)

  const float* Wg = gp_w + (size_t)e * INTER * HIDDEN;
  const float* Wu = up_w + (size_t)e * INTER * HIDDEN;

  __shared__ _Float16 lsA[2][BM * LDA];
  __shared__ _Float16 lsG[2][BN * LDA];
  __shared__ _Float16 lsU[2][BN * LDA];

  // A (f16): 128 rows x 32 halves = 512 x b128 chunks -> 2 per thread (async DMA)
  int             aDst[2];
  const _Float16* aSrc[2];
#pragma unroll
  for (int j = 0; j < 2; ++j) {
    const int c = tid + 256 * j;
    const int r = c >> 2;          // row 0..127
    const int seg = c & 3;         // 8-half segment
    int slot = mBase + r;
    if (slot >= count) slot = count - 1;   // clamp; epilogue masks writes
    aDst[j] = r * LDA + seg * 8;
    aSrc[j] = xh + (size_t)tok[slot] * HIDDEN + seg * 8;
  }
  // B (each of 2 matrices): 64 rows x 32 floats = 512 float4 chunks -> 2 per thread
  int    bDst[2];
  size_t bOff[2];
#pragma unroll
  for (int j = 0; j < 2; ++j) {
    const int c = tid + 256 * j;
    const int r = c >> 3;
    const int seg = c & 7;
    bDst[j] = r * LDA + seg * 4;
    bOff[j] = (size_t)(nBase + r) * HIDDEN + seg * 4;
  }

  v8f accG[2][2], accU[2][2];
  const v8f vzero = {};
#pragma unroll
  for (int i = 0; i < 2; ++i)
#pragma unroll
    for (int j = 0; j < 2; ++j) { accG[i][j] = vzero; accU[i][j] = vzero; }

  // ---- prologue: A-DMA into buf0, B regs for k=0
  float4 gReg[2], uReg[2];
#pragma unroll
  for (int j = 0; j < 2; ++j) {
    gReg[j] = *(const float4*)(Wg + bOff[j]);
    uReg[j] = *(const float4*)(Wu + bOff[j]);
    async_copy_b128(aSrc[j], &lsA[0][aDst[j]]);
  }

  int cur = 0;
  for (int k0 = 0; k0 < HIDDEN; k0 += BK) {
    const bool hasNext = (k0 + BK) < HIDDEN;
    // store B regs -> lsG/lsU[cur]
#pragma unroll
    for (int j = 0; j < 2; ++j) {
      Half4 hg, hu;
      hg.h[0] = (_Float16)gReg[j].x; hg.h[1] = (_Float16)gReg[j].y;
      hg.h[2] = (_Float16)gReg[j].z; hg.h[3] = (_Float16)gReg[j].w;
      hu.h[0] = (_Float16)uReg[j].x; hu.h[1] = (_Float16)uReg[j].y;
      hu.h[2] = (_Float16)uReg[j].z; hu.h[3] = (_Float16)uReg[j].w;
      *(uint2*)&lsG[cur][bDst[j]] = hg.u;
      *(uint2*)&lsU[cur][bDst[j]] = hu.u;
    }
    // register-prefetch B for next step
    float4 gN[2], uN[2];
    if (hasNext) {
#pragma unroll
      for (int j = 0; j < 2; ++j) {
        gN[j] = *(const float4*)(Wg + bOff[j] + k0 + BK);
        uN[j] = *(const float4*)(Wu + bOff[j] + k0 + BK);
      }
    }
    if (k0 + 8 * BK < HIDDEN) {           // L2 prefetch hints (global_prefetch_b8)
      __builtin_prefetch(Wg + bOff[0] + k0 + 8 * BK, 0, 0);
      __builtin_prefetch(Wu + bOff[0] + k0 + 8 * BK, 0, 0);
    }

    wait_async0();        // own A-DMAs into lsA[cur] retired
    __syncthreads();      // everyone's A(cur)/B(cur) visible; readers of cur^1 done

    // launch next A-DMA now: overlaps with the WMMAs below
    if (hasNext) {
#pragma unroll
      for (int j = 0; j < 2; ++j)
        async_copy_b128(aSrc[j] + k0 + BK, &lsA[cur ^ 1][aDst[j]]);
    }

    // Fragment loads per ISA 7.12.2 (16-bit A 16x32 / B 32x16 layouts)
    Frag16 a[2], bg[2], bu[2];
#pragma unroll
    for (int ms = 0; ms < 2; ++ms) {
      const int mrow = wm * 32 + ms * 16 + (lane & 15);
      const int kq = (lane < 16) ? 0 : 8;
      a[ms].q[0] = *(const uint4*)&lsA[cur][mrow * LDA + kq];
      a[ms].q[1] = *(const uint4*)&lsA[cur][mrow * LDA + kq + 16];
    }
#pragma unroll
    for (int ns = 0; ns < 2; ++ns) {
      const int nrow = wn * 32 + ns * 16 + (lane & 15);
      const int kb = (lane < 16) ? 0 : 8;
      bg[ns].q[0] = *(const uint4*)&lsG[cur][nrow * LDA + kb];
      bg[ns].q[1] = *(const uint4*)&lsG[cur][nrow * LDA + kb + 16];
      bu[ns].q[0] = *(const uint4*)&lsU[cur][nrow * LDA + kb];
      bu[ns].q[1] = *(const uint4*)&lsU[cur][nrow * LDA + kb + 16];
    }
#pragma unroll
    for (int ms = 0; ms < 2; ++ms)
#pragma unroll
      for (int ns = 0; ns < 2; ++ns) {
        accG[ms][ns] = __builtin_amdgcn_wmma_f32_16x16x32_f16(
            false, a[ms].h, false, bg[ns].h, (short)0, accG[ms][ns], false, false);
        accU[ms][ns] = __builtin_amdgcn_wmma_f32_16x16x32_f16(
            false, a[ms].h, false, bu[ns].h, (short)0, accU[ms][ns], false, false);
      }
    if (hasNext) {
#pragma unroll
      for (int j = 0; j < 2; ++j) { gReg[j] = gN[j]; uReg[j] = uN[j]; }
    }
    cur ^= 1;
  }

  // Epilogue: h = silu(g) * u via v_rcp. C layout: M = v + (lane>=16)*8, N = lane&15.
  const int mHi  = (lane >= 16) ? 8 : 0;
  const int nCol = lane & 15;
#pragma unroll
  for (int ms = 0; ms < 2; ++ms)
#pragma unroll
    for (int ns = 0; ns < 2; ++ns) {
      const int n = nBase + wn * 32 + ns * 16 + nCol;
#pragma unroll
      for (int v = 0; v < 8; ++v) {
        const int slot = mBase + wm * 32 + ms * 16 + mHi + v;
        if (slot < count) {
          const float g = accG[ms][ns][v];
          const float u = accU[ms][ns][v];
          const float hval = g * fast_rcp(1.0f + __expf(-g)) * u;
          hbuf[(size_t)slot * INTER + n] = (_Float16)hval;
        }
      }
    }
}

// ---------------------------------------------------------------------------
// 3) per-expert down projection: out[tok[slot]] += coef[slot] * (h * Dp^T)
//    A (f16 h-buffer) via async DMA; experts sequential -> plain RMW, no atomics.
// ---------------------------------------------------------------------------
__global__ __launch_bounds__(256)
void moe_down_kernel(const _Float16* __restrict__ hbuf,
                     const float* __restrict__ dp_w,
                     const int* __restrict__ counts,
                     const int* __restrict__ tok,    // already offset by expert
                     const float* __restrict__ cf,   // already offset by expert
                     float* __restrict__ out,
                     int e) {
  const int count = counts[e];
  const int mBase = blockIdx.y * BM;
  if (mBase >= count) return;
  const int nBase = blockIdx.x * BN;   // over HIDDEN

  const int tid  = threadIdx.x;
  const int lane = tid & 31;
  const int w    = tid >> 5;
  const int wm   = w & 3;
  const int wn   = w >> 2;

  const float* Wd = dp_w + (size_t)e * HIDDEN * INTER;

  __shared__ _Float16 lsA[2][BM * LDA];
  __shared__ _Float16 lsB[2][BN * LDA];

  int             aDst[2];
  const _Float16* aSrc[2];
#pragma unroll
  for (int j = 0; j < 2; ++j) {
    const int c = tid + 256 * j;
    const int r = c >> 2;
    const int seg = c & 3;
    int slot = mBase + r;
    if (slot >= count) slot = count - 1;
    aDst[j] = r * LDA + seg * 8;
    aSrc[j] = hbuf + (size_t)slot * INTER + seg * 8;
  }
  int    bDst[2];
  size_t bOff[2];
#pragma unroll
  for (int j = 0; j < 2; ++j) {
    const int c = tid + 256 * j;
    const int r = c >> 3;
    const int seg = c & 7;
    bDst[j] = r * LDA + seg * 4;
    bOff[j] = (size_t)(nBase + r) * INTER + seg * 4;
  }

  v8f acc[2][2];
  const v8f vzero = {};
#pragma unroll
  for (int i = 0; i < 2; ++i)
#pragma unroll
    for (int j = 0; j < 2; ++j) acc[i][j] = vzero;

  float4 bReg[2];
#pragma unroll
  for (int j = 0; j < 2; ++j) {
    bReg[j] = *(const float4*)(Wd + bOff[j]);
    async_copy_b128(aSrc[j], &lsA[0][aDst[j]]);
  }

  int cur = 0;
  for (int k0 = 0; k0 < INTER; k0 += BK) {
    const bool hasNext = (k0 + BK) < INTER;
#pragma unroll
    for (int j = 0; j < 2; ++j) {
      Half4 hv;
      hv.h[0] = (_Float16)bReg[j].x; hv.h[1] = (_Float16)bReg[j].y;
      hv.h[2] = (_Float16)bReg[j].z; hv.h[3] = (_Float16)bReg[j].w;
      *(uint2*)&lsB[cur][bDst[j]] = hv.u;
    }
    float4 bN[2];
    if (hasNext) {
#pragma unroll
      for (int j = 0; j < 2; ++j)
        bN[j] = *(const float4*)(Wd + bOff[j] + k0 + BK);
    }
    if (k0 + 8 * BK < INTER)
      __builtin_prefetch(Wd + bOff[0] + k0 + 8 * BK, 0, 0);

    wait_async0();
    __syncthreads();

    if (hasNext) {
#pragma unroll
      for (int j = 0; j < 2; ++j)
        async_copy_b128(aSrc[j] + k0 + BK, &lsA[cur ^ 1][aDst[j]]);
    }

    Frag16 a[2], b[2];
#pragma unroll
    for (int ms = 0; ms < 2; ++ms) {
      const int mrow = wm * 32 + ms * 16 + (lane & 15);
      const int kq = (lane < 16) ? 0 : 8;
      a[ms].q[0] = *(const uint4*)&lsA[cur][mrow * LDA + kq];
      a[ms].q[1] = *(const uint4*)&lsA[cur][mrow * LDA + kq + 16];
    }
#pragma unroll
    for (int ns = 0; ns < 2; ++ns) {
      const int nrow = wn * 32 + ns * 16 + (lane & 15);
      const int kb = (lane < 16) ? 0 : 8;
      b[ns].q[0] = *(const uint4*)&lsB[cur][nrow * LDA + kb];
      b[ns].q[1] = *(const uint4*)&lsB[cur][nrow * LDA + kb + 16];
    }
#pragma unroll
    for (int ms = 0; ms < 2; ++ms)
#pragma unroll
      for (int ns = 0; ns < 2; ++ns)
        acc[ms][ns] = __builtin_amdgcn_wmma_f32_16x16x32_f16(
            false, a[ms].h, false, b[ns].h, (short)0, acc[ms][ns], false, false);
    if (hasNext) {
#pragma unroll
      for (int j = 0; j < 2; ++j) bReg[j] = bN[j];
    }
    cur ^= 1;
  }

  const int mHi  = (lane >= 16) ? 8 : 0;
  const int nCol = lane & 15;
#pragma unroll
  for (int ms = 0; ms < 2; ++ms)
#pragma unroll
    for (int ns = 0; ns < 2; ++ns) {
      const int n = nBase + wn * 32 + ns * 16 + nCol;
#pragma unroll
      for (int v = 0; v < 8; ++v) {
        const int slot = mBase + wm * 32 + ms * 16 + mHi + v;
        if (slot < count) {
          const int t = tok[slot];
          float* o = out + (size_t)t * HIDDEN + n;
          *o = *o + cf[slot] * acc[ms][ns][v];
        }
      }
    }
}

// ---------------------------------------------------------------------------
// launch
// ---------------------------------------------------------------------------
extern "C" void kernel_launch(void* const* d_in, const int* in_sizes, int n_in,
                              void* d_out, int out_size, void* d_ws, size_t ws_size,
                              hipStream_t stream) {
  (void)in_sizes; (void)n_in; (void)out_size; (void)ws_size;

  const float* x      = (const float*)d_in[0];
  const float* gate_w = (const float*)d_in[1];
  const float* gp_w   = (const float*)d_in[2];
  const float* up_w   = (const float*)d_in[3];
  const float* dp_w   = (const float*)d_in[4];
  float* out = (float*)d_out;

  char* ws = (char*)d_ws;
  constexpr size_t WS_TOK  = 1024;
  constexpr size_t WS_COEF = WS_TOK + (size_t)NEXP * NTOK * sizeof(int);
  constexpr size_t WS_XH   = WS_COEF + (size_t)NEXP * NTOK * sizeof(float);
  constexpr size_t WS_HBUF = WS_XH + (size_t)NTOK * HIDDEN * 2;
  int*      counts   = (int*)(ws + 0);
  int*      tok_list = (int*)(ws + WS_TOK);
  float*    coefs    = (float*)(ws + WS_COEF);
  _Float16* xh       = (_Float16*)(ws + WS_XH);     // NTOK*HIDDEN halves (16 MB)
  _Float16* hbuf     = (_Float16*)(ws + WS_HBUF);   // NTOK*INTER halves (~44 MB)

  moe_zero_kernel<<<2048, 256, 0, stream>>>(out, counts);
  moe_xcast_kernel<<<2048, 256, 0, stream>>>(x, xh);
  moe_router_kernel<<<NTOK / 8, 256, 0, stream>>>(x, gate_w, counts, tok_list, coefs);

  const dim3 ffnGrid(INTER / BN, NTOK / BM);    // 88 x 32, blocks early-exit past count
  const dim3 dnGrid(HIDDEN / BN, NTOK / BM);    // 32 x 32
  for (int e = 0; e < NEXP; ++e) {
    moe_ffn_kernel<<<ffnGrid, 256, 0, stream>>>(
        xh, gp_w, up_w, counts, tok_list + e * NTOK, hbuf, e);
    moe_down_kernel<<<dnGrid, 256, 0, stream>>>(
        hbuf, dp_w, counts, tok_list + e * NTOK, coefs + e * NTOK, out, e);
  }
}
